// SegmentVQVAE_70351564308896
// MI455X (gfx1250) — matrix operand order repaired and checked by reading, not compiled, not run on hardware
//
#include <hip/hip_runtime.h>
#include <hip/hip_bf16.h>
#include <stdint.h>

typedef __attribute__((ext_vector_type(16))) __bf16 v16bf;
typedef __attribute__((ext_vector_type(8)))  float  v8f;
typedef __attribute__((__vector_size__(16))) int    i32x4;

#define B_      128
#define T_      100
#define TP_     102   // padded T
#define C_      4
#define EMB_    128
#define HID_    512
#define LAT_    256
#define K_CODES 1024
#define P_      8

#define WMMA_BF16(a, b, c) \
    __builtin_amdgcn_wmma_f32_16x16x32_bf16(false, (a), false, (b), (short)0, (c), false, false)

#if __has_builtin(__builtin_amdgcn_global_load_async_to_lds_b128)
#define HAVE_ASYNC_LDS 1
#else
#define HAVE_ASYNC_LDS 0
#endif

// global (AS1) / LDS (AS3) pointer casts with the builtin's exact int4-vector type
#define GPTR128(p) ((__attribute__((address_space(1))) i32x4*)(uintptr_t)(p))
#define LPTR128(p) ((__attribute__((address_space(3))) i32x4*)(uint32_t)(uintptr_t)(p))

__device__ __forceinline__ void wait_async0() {
#if __has_builtin(__builtin_amdgcn_s_wait_asynccnt)
    __builtin_amdgcn_s_wait_asynccnt(0);
#else
    asm volatile("s_wait_asynccnt 0x0" ::: "memory");
#endif
}

// ---------------------------------------------------------------------------
// Generic bf16 WMMA GEMM:  C[m,n] = act( sum_k A[m,k]*W[n,k] + bias[n] )
// A row address:  ((m/aRowsPer)*aBatchStride + m%aRowsPer)*lda + k   (im2col)
// C row address:  ((m/oRowsPer)*oBatchStride + m%oRowsPer + oOff)*ldc + n
// Block tile 128x64, K-step 32, 256 threads = 8 waves, 2x2 WMMA tiles / wave.
// Double-buffered LDS; async global->LDS copies when available.
// ---------------------------------------------------------------------------
__global__ __launch_bounds__(256) void gemm_bf16_kernel(
    const __bf16* __restrict__ A, const __bf16* __restrict__ W,
    const float* __restrict__ bias, void* __restrict__ Cout,
    int M, int N, int K,
    int aRowsPer, int aBatchStride, int lda,
    int oRowsPer, int oBatchStride, int oOff, int ldc,
    int doRelu, int outBf16)
{
    __shared__ __bf16 sA[2][128][32];   // 16 KB
    __shared__ __bf16 sB[2][64][32];    //  8 KB

    const int bm = blockIdx.y * 128;
    const int bn = blockIdx.x * 64;
    const int tid  = threadIdx.x;
    const int wave = tid >> 5;
    const int lane = tid & 31;
    const int half = lane >> 4;        // 0..1
    const int l16  = lane & 15;        // 0..15
    const int mt0  = (wave >> 1) * 2;  // 0,2,4,6
    const int nt0  = (wave & 1) * 2;   // 0,2

    // loader mapping: A = two 16B chunks/thread (rows r, r+64); B = one chunk
    const int rA0 = tid >> 2;          // 0..63
    const int rA1 = rA0 + 64;          // 64..127
    const int cL  = (tid & 3) * 8;     // 0,8,16,24

    const int gA0 = bm + rA0, gA1 = bm + rA1;
    const long aOff0 = ((long)(gA0 / aRowsPer) * aBatchStride + (gA0 % aRowsPer)) * (long)lda + cL;
    const long aOff1 = ((long)(gA1 / aRowsPer) * aBatchStride + (gA1 % aRowsPer)) * (long)lda + cL;
    const long bOff  = (long)(bn + rA0) * (long)K + cL;

    v8f acc00 = {}, acc01 = {}, acc10 = {}, acc11 = {};

    const int S = K >> 5;

#if HAVE_ASYNC_LDS
    auto stage = [&](int buf, int kk) {
        __builtin_amdgcn_global_load_async_to_lds_b128(
            GPTR128(A + aOff0 + kk), LPTR128(&sA[buf][rA0][cL]), 0, 0);
        __builtin_amdgcn_global_load_async_to_lds_b128(
            GPTR128(A + aOff1 + kk), LPTR128(&sA[buf][rA1][cL]), 0, 0);
        __builtin_amdgcn_global_load_async_to_lds_b128(
            GPTR128(W + bOff + kk), LPTR128(&sB[buf][rA0][cL]), 0, 0);
    };
#else
    float4 ra0, ra1, rb;
    auto regload = [&](int kk) {
        ra0 = *(const float4*)(A + aOff0 + kk);
        ra1 = *(const float4*)(A + aOff1 + kk);
        rb  = *(const float4*)(W + bOff + kk);
    };
    auto stash = [&](int buf) {
        *(float4*)&sA[buf][rA0][cL] = ra0;
        *(float4*)&sA[buf][rA1][cL] = ra1;
        *(float4*)&sB[buf][rA0][cL] = rb;
    };
#endif

    auto compute = [&](int cur) {
        // B fragment: elem e -> (K = half*16 + e, N = l16)
        v16bf b0 = *(const v16bf*)&sB[cur][nt0 * 16 + l16][half * 16];
        v16bf b1 = *(const v16bf*)&sB[cur][(nt0 + 1) * 16 + l16][half * 16];
        // A fragment: elem e -> K = half*16 + e
        v16bf a0, a1;
        *((float4*)&a0)       = *(const float4*)&sA[cur][mt0 * 16 + l16][half * 16];
        *(((float4*)&a0) + 1) = *(const float4*)&sA[cur][mt0 * 16 + l16][half * 16 + 8];
        *((float4*)&a1)       = *(const float4*)&sA[cur][(mt0 + 1) * 16 + l16][half * 16];
        *(((float4*)&a1) + 1) = *(const float4*)&sA[cur][(mt0 + 1) * 16 + l16][half * 16 + 8];
        acc00 = WMMA_BF16(a0, b0, acc00);
        acc01 = WMMA_BF16(a0, b1, acc01);
        acc10 = WMMA_BF16(a1, b0, acc10);
        acc11 = WMMA_BF16(a1, b1, acc11);
    };

#if HAVE_ASYNC_LDS
    stage(0, 0);
    wait_async0();
    __syncthreads();
    for (int s = 0; s < S; ++s) {
        const int cur = s & 1;
        if (s + 1 < S) stage(cur ^ 1, (s + 1) << 5);
        compute(cur);
        if (s + 1 < S) wait_async0();
        __syncthreads();
    }
#else
    regload(0);
    stash(0);
    __syncthreads();
    for (int s = 0; s < S; ++s) {
        const int cur = s & 1;
        if (s + 1 < S) regload((s + 1) << 5);
        compute(cur);
        if (s + 1 < S) stash(cur ^ 1);
        __syncthreads();
    }
#endif

    for (int ti = 0; ti < 2; ++ti) {
        for (int tj = 0; tj < 2; ++tj) {
            v8f acc = (ti == 0) ? (tj == 0 ? acc00 : acc01) : (tj == 0 ? acc10 : acc11);
            const int n = bn + (nt0 + tj) * 16 + l16;
            const float bv = bias ? bias[n] : 0.0f;
            const int mBase = bm + (mt0 + ti) * 16 + 8 * half;  // C layout: M = r + 8*(lane/16)
            for (int r = 0; r < 8; ++r) {
                int m = mBase + r;
                float v = acc[r] + bv;
                if (doRelu) v = fmaxf(v, 0.0f);
                long off = ((long)(m / oRowsPer) * oBatchStride + (m % oRowsPer) + oOff) * (long)ldc + n;
                if (outBf16) ((__bf16*)Cout)[off] = (__bf16)v;
                else         ((float*)Cout)[off]  = v;
            }
        }
    }
}

// ---------------------------------------------------------------------------
// Small helper kernels
// ---------------------------------------------------------------------------
__global__ void zero_f32_kernel(float* p, int n) {
    int i = blockIdx.x * blockDim.x + threadIdx.x;
    if (i < n) p[i] = 0.0f;
}

__global__ void zero_pad_rows_kernel(__bf16* buf) {   // zero rows 0 and 101 of each batch block
    int i = blockIdx.x * blockDim.x + threadIdx.x;    // B_*2*HID_
    if (i >= B_ * 2 * HID_) return;
    int e = i % HID_, which = (i / HID_) % 2, b = i / (2 * HID_);
    buf[((long)b * TP_ + which * (TP_ - 1)) * HID_ + e] = (__bf16)0.0f;
}

__global__ void embed_kernel(const int* __restrict__ tok, const float* __restrict__ emb,
                             __bf16* __restrict__ Xpad) {
    int i = blockIdx.x * blockDim.x + threadIdx.x;    // B_*T_*HID_
    if (i >= B_ * T_ * HID_) return;
    int e = i % HID_;
    int t = (i / HID_) % T_;
    int b = i / (HID_ * T_);
    int cb = e / EMB_, ee = e % EMB_;
    int tk = tok[(b * T_ + t) * C_ + cb];
    Xpad[((long)b * TP_ + t + 1) * HID_ + e] = (__bf16)emb[tk * EMB_ + ee];
}

// conv weight (O, Cin, 3) -> bf16 (O, 3*Cin) with k index = tap*Cin + c
__global__ void prep_conv_w_kernel(const float* __restrict__ w, __bf16* __restrict__ out, int Cin) {
    int i = blockIdx.x * blockDim.x + threadIdx.x;
    int tot = HID_ * Cin * 3;
    if (i >= tot) return;
    int o = i / (3 * Cin);
    int r = i % (3 * Cin);
    int tap = r / Cin, c = r % Cin;
    out[i] = (__bf16)w[(o * Cin + c) * 3 + tap];
}

__global__ void cast_bf16_kernel(const float* __restrict__ in, __bf16* __restrict__ out, int n) {
    int i = blockIdx.x * blockDim.x + threadIdx.x;
    if (i < n) out[i] = (__bf16)in[i];
}

// Fold deconv1 (ConvTranspose k=3,pad=1 applied to T-constant input) into 3 matrices.
// d1w layout (Cin=512, Cout=512, 3).  cls0: t=0 (taps 0+1), cls1: interior (0+1+2), cls2: t=99 (1+2)
__global__ void prep_d1_kernel(const float* __restrict__ d1w, const float* __restrict__ d1b,
                               __bf16* __restrict__ Wcat, float* __restrict__ biascat) {
    int i = blockIdx.x * blockDim.x + threadIdx.x;    // 1536*512
    if (i >= 3 * HID_ * HID_) return;
    int n = i / HID_, kin = i % HID_;
    int cls = n / HID_, o = n % HID_;
    const float* base = d1w + ((long)kin * HID_ + o) * 3;
    float v;
    if (cls == 0)      v = base[0] + base[1];
    else if (cls == 1) v = base[0] + base[1] + base[2];
    else               v = base[1] + base[2];
    Wcat[i] = (__bf16)v;
    if (i < 3 * HID_) biascat[i] = d1b[i % HID_];
}

// deconv2_w (512, 4096) -> transposed bf16 (4096, 512)
__global__ void prep_d2_kernel(const float* __restrict__ d2w, __bf16* __restrict__ Wt) {
    int i = blockIdx.x * blockDim.x + threadIdx.x;    // 4096*512
    if (i >= 4096 * HID_) return;
    int n = i / HID_, k = i % HID_;
    Wt[i] = (__bf16)d2w[(long)k * 4096 + n];
}

// adaptive avg pool: Y2 (B,100,512) f32 -> pooled (B,8,512)
__global__ void pool_kernel(const float* __restrict__ Y2, float* __restrict__ pooled) {
    int i = blockIdx.x * blockDim.x + threadIdx.x;    // B_*P_*HID_
    if (i >= B_ * P_ * HID_) return;
    int c = i % HID_;
    int p = (i / HID_) % P_;
    int b = i / (HID_ * P_);
    int st = (p * T_) / P_;
    int en = ((p + 1) * T_ + P_ - 1) / P_;
    float s = 0.0f;
    for (int t = st; t < en; ++t) s += Y2[((long)b * T_ + t) * HID_ + c];
    pooled[i] = s / (float)(en - st);
}

// fused 1x1 conv3 + VQ (argmin + gather + commit accumulation); one block per (b,p)
__global__ __launch_bounds__(256) void conv3_vq_kernel(
    const float* __restrict__ pooled, const float* __restrict__ w3, const float* __restrict__ b3,
    const float* __restrict__ codebook, __bf16* __restrict__ zcat, int seg, float* __restrict__ accum)
{
    int b = blockIdx.x / P_, p = blockIdx.x % P_;
    __shared__ float sIn[HID_];
    __shared__ float sZ[LAT_];
    __shared__ float sRed[256];
    __shared__ int   sIdx[256];
    int tid = threadIdx.x;
    sIn[tid]       = pooled[((long)b * P_ + p) * HID_ + tid];
    sIn[tid + 256] = pooled[((long)b * P_ + p) * HID_ + 256 + tid];
    __syncthreads();
    if (tid < LAT_) {
        float z = b3[tid];
        const float* wr = w3 + (long)tid * HID_;
        for (int c = 0; c < HID_; ++c) z += sIn[c] * wr[c];
        sZ[tid] = z;
    }
    __syncthreads();
    float best = 3.4e38f; int bestk = 0;
    for (int kk = tid; kk < K_CODES; kk += 256) {
        const float* cb = codebook + (long)kk * LAT_;
        float d = 0.0f;
        for (int c = 0; c < LAT_; ++c) { float t = sZ[c] - cb[c]; d += t * t; }
        if (d < best) { best = d; bestk = kk; }
    }
    sRed[tid] = best; sIdx[tid] = bestk;
    __syncthreads();
    for (int s = 128; s > 0; s >>= 1) {
        if (tid < s && sRed[tid + s] < sRed[tid]) { sRed[tid] = sRed[tid + s]; sIdx[tid] = sIdx[tid + s]; }
        __syncthreads();
    }
    int code = sIdx[0];
    __syncthreads();
    float part = 0.0f;
    if (tid < LAT_) {
        float cq = codebook[(long)code * LAT_ + tid];
        zcat[(long)b * (3 * P_ * LAT_) + seg * (P_ * LAT_) + p * LAT_ + tid] = (__bf16)cq;
        float df = sZ[tid] - cq;
        part = df * df;
    }
    sRed[tid] = part;
    __syncthreads();
    for (int s = 128; s > 0; s >>= 1) {
        if (tid < s) sRed[tid] += sRed[tid + s];
        __syncthreads();
    }
    if (tid == 0) atomicAdd(&accum[1], sRed[0]);
}

// log-sum-exp over groups of 1024 logits; one block per (row, codebook)
__global__ __launch_bounds__(256) void lse_kernel(const float* __restrict__ logits, float* __restrict__ lse) {
    int row = blockIdx.x / C_, c = blockIdx.x % C_;
    const float* p = logits + (long)row * 4096 + c * 1024;
    __shared__ float red[256];
    int tid = threadIdx.x;
    float mx = -3.4e38f;
    for (int v = tid; v < 1024; v += 256) mx = fmaxf(mx, p[v]);
    red[tid] = mx; __syncthreads();
    for (int s = 128; s > 0; s >>= 1) { if (tid < s) red[tid] = fmaxf(red[tid], red[tid + s]); __syncthreads(); }
    mx = red[0]; __syncthreads();
    float sum = 0.0f;
    for (int v = tid; v < 1024; v += 256) sum += __expf(p[v] - mx);
    red[tid] = sum; __syncthreads();
    for (int s = 128; s > 0; s >>= 1) { if (tid < s) red[tid] += red[tid + s]; __syncthreads(); }
    if (tid == 0) lse[blockIdx.x] = mx + __logf(red[0]);
}

__global__ __launch_bounds__(256) void nll_kernel(const int* __restrict__ tokens_curr,
                                                  const float* __restrict__ logits,
                                                  const float* __restrict__ lse,
                                                  float* __restrict__ accum) {
    int idx = blockIdx.x * 256 + threadIdx.x;         // B_*T_*C_ = 51200
    float val = 0.0f;
    if (idx < B_ * T_ * C_) {
        int c = idx % C_;
        int t = (idx / C_) % T_;
        int b = idx / (C_ * T_);
        int tk = tokens_curr[idx];
        int cls = (t == 0) ? 0 : (t == T_ - 1) ? 2 : 1;
        int row = b * 3 + cls;
        val = lse[row * C_ + c] - logits[(long)row * 4096 + c * 1024 + tk];
    }
    __shared__ float red[256];
    int tid = threadIdx.x;
    red[tid] = val; __syncthreads();
    for (int s = 128; s > 0; s >>= 1) { if (tid < s) red[tid] += red[tid + s]; __syncthreads(); }
    if (tid == 0) atomicAdd(&accum[0], red[0]);
}

__global__ void finalize_kernel(const float* __restrict__ accum, float* __restrict__ out) {
    out[0] = accum[0] / (float)(B_ * T_ * C_) + 0.1f * (accum[1] / (float)(B_ * P_ * LAT_));
}

// ---------------------------------------------------------------------------
// Host side
// ---------------------------------------------------------------------------
static void gemm(hipStream_t s, const __bf16* A, const __bf16* W, const float* bias, void* C,
                 int M, int N, int K, int aRowsPer, int aBatch, int lda,
                 int oRowsPer, int oBatch, int oOff, int ldc, int relu, int obf16) {
    dim3 g(N / 64, M / 128);
    gemm_bf16_kernel<<<g, 256, 0, s>>>(A, W, bias, C, M, N, K,
                                       aRowsPer, aBatch, lda, oRowsPer, oBatch, oOff, ldc, relu, obf16);
}

extern "C" void kernel_launch(void* const* d_in, const int* in_sizes, int n_in,
                              void* d_out, int out_size, void* d_ws, size_t ws_size,
                              hipStream_t stream) {
    const int* tokens[3] = { (const int*)d_in[0], (const int*)d_in[1], (const int*)d_in[2] };
    const float* emb       = (const float*)d_in[3];
    const float* conv1_w   = (const float*)d_in[4];
    const float* conv1_b   = (const float*)d_in[5];
    const float* conv2_w   = (const float*)d_in[6];
    const float* conv2_b   = (const float*)d_in[7];
    const float* conv3_w   = (const float*)d_in[8];
    const float* conv3_b   = (const float*)d_in[9];
    const float* codebook  = (const float*)d_in[10];
    const float* fc1_w     = (const float*)d_in[11];
    const float* fc1_b     = (const float*)d_in[12];
    const float* fc2_w     = (const float*)d_in[13];
    const float* fc2_b     = (const float*)d_in[14];
    const float* deconv1_w = (const float*)d_in[15];
    const float* deconv1_b = (const float*)d_in[16];
    const float* deconv2_w = (const float*)d_in[17];
    const float* deconv2_b = (const float*)d_in[18];

    // workspace carve
    char* w = (char*)d_ws;
    auto carve = [&](size_t bytes) { void* p = (void*)w; w += (bytes + 255) & ~(size_t)255; return p; };
    float*  accum   = (float*)  carve(256);
    __bf16* Xpad    = (__bf16*) carve((size_t)B_ * TP_ * HID_ * 2);
    __bf16* Y1pad   = (__bf16*) carve((size_t)B_ * TP_ * HID_ * 2);
    float*  Y2      = (float*)  carve((size_t)B_ * T_ * HID_ * 4);
    float*  pooled  = (float*)  carve((size_t)B_ * P_ * HID_ * 4);
    __bf16* zcat    = (__bf16*) carve((size_t)B_ * 3 * P_ * LAT_ * 2);
    __bf16* h1      = (__bf16*) carve((size_t)B_ * HID_ * 2);
    __bf16* h2      = (__bf16*) carve((size_t)B_ * HID_ * 2);
    __bf16* Wc1     = (__bf16*) carve((size_t)HID_ * 1536 * 2);
    __bf16* Wc2     = (__bf16*) carve((size_t)HID_ * 1536 * 2);
    __bf16* Wfc1    = (__bf16*) carve((size_t)HID_ * 6144 * 2);
    __bf16* Wfc2    = (__bf16*) carve((size_t)HID_ * HID_ * 2);
    __bf16* Wd1     = (__bf16*) carve((size_t)1536 * HID_ * 2);
    float*  biascat = (float*)  carve(1536 * 4);
    __bf16* Wd2     = (__bf16*) carve((size_t)4096 * HID_ * 2);
    __bf16* H3      = (__bf16*) carve((size_t)B_ * 1536 * 2);
    float*  logits  = (float*)  carve((size_t)B_ * 3 * 4096 * 4);
    float*  lseb    = (float*)  carve((size_t)B_ * 3 * C_ * 4);
    if ((size_t)(w - (char*)d_ws) > ws_size) return;

    const int TB = 256;
    auto nb = [](int n) { return (n + 255) / 256; };

    zero_f32_kernel<<<1, 8, 0, stream>>>(accum, 8);

    // weight prep (f32 -> bf16 with required permutations / foldings)
    prep_conv_w_kernel<<<nb(HID_ * 1536), TB, 0, stream>>>(conv1_w, Wc1, 512);
    prep_conv_w_kernel<<<nb(HID_ * 1536), TB, 0, stream>>>(conv2_w, Wc2, 512);
    cast_bf16_kernel<<<nb(HID_ * 6144), TB, 0, stream>>>(fc1_w, Wfc1, HID_ * 6144);
    cast_bf16_kernel<<<nb(HID_ * HID_), TB, 0, stream>>>(fc2_w, Wfc2, HID_ * HID_);
    prep_d1_kernel<<<nb(1536 * HID_), TB, 0, stream>>>(deconv1_w, deconv1_b, Wd1, biascat);
    prep_d2_kernel<<<nb(4096 * HID_), TB, 0, stream>>>(deconv2_w, Wd2);

    // encoder + VQ per segment (shared scratch reused sequentially)
    for (int seg = 0; seg < 3; ++seg) {
        zero_pad_rows_kernel<<<nb(B_ * 2 * HID_), TB, 0, stream>>>(Xpad);
        zero_pad_rows_kernel<<<nb(B_ * 2 * HID_), TB, 0, stream>>>(Y1pad);
        embed_kernel<<<nb(B_ * T_ * HID_), TB, 0, stream>>>(tokens[seg], emb, Xpad);
        // conv1: im2col GEMM (12800 x 1536) x (1536 x 512) -> padded bf16 Y1
        gemm(stream, Xpad, Wc1, conv1_b, Y1pad, B_ * T_, HID_, 1536,
             T_, TP_, HID_, T_, TP_, 1, HID_, 1, 1);
        // conv2: same shape -> f32 Y2
        gemm(stream, Y1pad, Wc2, conv2_b, Y2, B_ * T_, HID_, 1536,
             T_, TP_, HID_, B_ * T_, 0, 0, HID_, 1, 0);
        pool_kernel<<<nb(B_ * P_ * HID_), TB, 0, stream>>>(Y2, pooled);
        conv3_vq_kernel<<<B_ * P_, 256, 0, stream>>>(pooled, conv3_w, conv3_b, codebook, zcat, seg, accum);
    }

    // decoder
    gemm(stream, zcat, Wfc1, fc1_b, h1, B_, HID_, 6144, B_, 0, 6144, B_, 0, 0, HID_, 1, 1);
    gemm(stream, h1, Wfc2, fc2_b, h2, B_, HID_, HID_, B_, 0, HID_, B_, 0, 0, HID_, 1, 1);
    // deconv1 folded into 3 t-class matmuls -> H3 (128 x 1536)
    gemm(stream, h2, Wd1, biascat, H3, B_, 1536, HID_, B_, 0, HID_, B_, 0, 0, 1536, 1, 1);
    // deconv2 on 384 distinct (b, t-class) rows -> logits (384 x 4096)
    gemm(stream, H3, Wd2, deconv2_b, logits, B_ * 3, 4096, HID_,
         B_ * 3, 0, HID_, B_ * 3, 0, 0, 4096, 0, 0);

    // loss
    lse_kernel<<<B_ * 3 * C_, 256, 0, stream>>>(logits, lseb);
    nll_kernel<<<nb(B_ * T_ * C_), 256, 0, stream>>>(tokens[1], logits, lseb, accum);
    finalize_kernel<<<1, 1, 0, stream>>>(accum, (float*)d_out);
}
